// SRGNN_3040836846451
// MI455X (gfx1250) — matrix operand order repaired and checked by reading, not compiled, not run on hardware
//
#include <hip/hip_runtime.h>

// ---------------------------------------------------------------------------
// SRGNN session scoring, MI455X (gfx1250, wave32, WMMA).
//
// Pipeline (all on `stream`, serialized):
//   k_conv_emb : item_emb f32 -> (hi, lo) bf16 planes        (bandwidth, ~9us)
//   k_trans_wg : W_g -> W_g^T (hi, lo) bf16 planes           (tiny)
//   k_local    : local = emb[x[:, -1]] @ W_l + b_l           (VALU, tiny)
//   k_attn     : fused global proj (split-bf16 WMMA) + additive attention
//                + weighted pooling -> session_global        (WMMA, ~5us)
//   k_hybrid   : [local|global] @ W_h + b_h -> bf16 hi/lo    (VALU, tiny)
//   k_score    : hybrid @ emb.T via split-bf16 WMMA          (BW-bound, ~18us)
//
// Split-bf16: x = hi + lo, hi = bf16(x), lo = bf16(x - hi).
// A@B ~= Ahi@Bhi + Ahi@Blo + Alo@Bhi  (3x v_wmma_f32_16x16x32_bf16,
// rel. error ~2^-17, near-f32 accuracy at bf16-WMMA speed).
// ---------------------------------------------------------------------------

#define B_DIM   1024
#define S_DIM   50
#define D_DIM   256
#define N_ITEMS 100000
#define NTILES_N (N_ITEMS / 16)   // 6250, exact

typedef __attribute__((ext_vector_type(16))) __bf16 bf16x16;
typedef __attribute__((ext_vector_type(8)))  __bf16 bf16x8;
typedef __attribute__((ext_vector_type(4)))  __bf16 bf16x4;
typedef __attribute__((ext_vector_type(8)))  float  f32x8;

// Build a 16-element bf16 fragment from two 8-element (16B) runs.
// A-frag runs: [k0 + half*16, +16) contiguous               (ptr, ptr+8)
// B-frag runs: [k0 + half*16, +16) contiguous               (ptr, ptr+8)
__device__ __forceinline__ bf16x16 frag_ld(const __bf16* lo, const __bf16* hi) {
  bf16x8 a = *(const bf16x8*)lo;
  bf16x8 b = *(const bf16x8*)hi;
  return __builtin_shufflevector(a, b, 0, 1, 2, 3, 4, 5, 6, 7,
                                       8, 9, 10, 11, 12, 13, 14, 15);
}

__device__ __forceinline__ f32x8 wmma_bf16(bf16x16 a, bf16x16 b, f32x8 c) {
  // (neg_a, A, neg_b, B, c_mod, C, reuse_a, reuse_b)
  return __builtin_amdgcn_wmma_f32_16x16x32_bf16(false, a, false, b,
                                                 (short)0, c, false, false);
}

// ---------------------------------------------------------------------------
// f32 -> (hi, lo) bf16 planes for item_emb, vectorized float4.
// ---------------------------------------------------------------------------
__global__ void k_conv_emb(const float* __restrict__ emb,
                           __bf16* __restrict__ h, __bf16* __restrict__ l,
                           int n4) {
  int stride = gridDim.x * blockDim.x;
  for (int i = blockIdx.x * blockDim.x + threadIdx.x; i < n4; i += stride) {
    float4 v = ((const float4*)emb)[i];
    bf16x4 vh, vl;
    vh.x = (__bf16)v.x; vl.x = (__bf16)(v.x - (float)vh.x);
    vh.y = (__bf16)v.y; vl.y = (__bf16)(v.y - (float)vh.y);
    vh.z = (__bf16)v.z; vl.z = (__bf16)(v.z - (float)vh.z);
    vh.w = (__bf16)v.w; vl.w = (__bf16)(v.w - (float)vh.w);
    ((bf16x4*)h)[i] = vh;
    ((bf16x4*)l)[i] = vl;
  }
}

// W_g (K-major, DxD) -> transposed bf16 hi/lo planes so WMMA B-fragments
// become contiguous b128 loads. Tiny (256 KB), uncoalesced reads are fine.
__global__ void k_trans_wg(const float* __restrict__ Wg,
                           __bf16* __restrict__ th, __bf16* __restrict__ tl) {
  int n = blockIdx.x, k = threadIdx.x;
  float v = Wg[k * D_DIM + n];
  __bf16 h = (__bf16)v;
  th[n * D_DIM + k] = h;
  tl[n * D_DIM + k] = (__bf16)(v - (float)h);
}

// ---------------------------------------------------------------------------
// local = emb[x[b, S-1]] @ W_l + b_l  (exact f32; 0.13 GFLOP, VALU)
// One block per b, thread d accumulates a column; W_l rows stream coalesced.
// ---------------------------------------------------------------------------
__global__ void k_local(const int* __restrict__ x, const float* __restrict__ emb,
                        const float* __restrict__ Wl, const float* __restrict__ bl,
                        float* __restrict__ localf) {
  int b = blockIdx.x, d = threadIdx.x;
  int idx = x[b * S_DIM + (S_DIM - 1)];
  const float* e = emb + (size_t)idx * D_DIM;
  float acc = bl[d];
#pragma unroll 4
  for (int k = 0; k < D_DIM; ++k)
    acc = fmaf(e[k], Wl[k * D_DIM + d], acc);
  localf[b * D_DIM + d] = acc;
}

// ---------------------------------------------------------------------------
// Fused attention per batch element.
//   G = gather(emb, x[b]) @ W_g          (split-bf16 WMMA, M=64 padded, K=256)
//   score[s] = sigmoid(G + local + bias) @ W_q + b_q   (shuffle reduction)
//   sess[b]  = sum_s emb[x[b,s]] * score[s]            (exact f32 pooling)
// M is processed in two 32-row passes to keep static LDS < 64 KB.
// ---------------------------------------------------------------------------
#define PITCH 264   // 256 + 8 bf16 pad -> row stride 528B, bank-conflict-free
__global__ void __launch_bounds__(256) k_attn(
    const int* __restrict__ x, const float* __restrict__ emb,
    const __bf16* __restrict__ WgTh, const __bf16* __restrict__ WgTl,
    const float* __restrict__ bg, const float* __restrict__ localf,
    const float* __restrict__ bias, const float* __restrict__ Wq,
    const float* __restrict__ bq, float* __restrict__ sess) {
  __shared__ __bf16 Ah[32 * PITCH];
  __shared__ __bf16 Al[32 * PITCH];
  __shared__ float part[64][16];
  __shared__ float scoreS[64];

  int b = blockIdx.x;
  int tid = threadIdx.x;
  int wave = tid >> 5, lane = tid & 31, col = lane & 15, half = lane >> 4;
  float biasb = bias[b];

  for (int p = 0; p < 2; ++p) {
    // ---- stage rows p*32 .. p*32+31 into LDS as bf16 hi/lo (zeros past S)
    for (int lin = tid; lin < 32 * D_DIM; lin += 256) {
      int rr = lin >> 8, c = lin & 255;
      int r = p * 32 + rr;
      float v = 0.0f;
      if (r < S_DIM) v = emb[(size_t)x[b * S_DIM + r] * D_DIM + c];
      __bf16 h = (__bf16)v;
      Ah[rr * PITCH + c] = h;
      Al[rr * PITCH + c] = (__bf16)(v - (float)h);
    }
    __syncthreads();

    // ---- 32 tile jobs (2 M-tiles x 16 N-tiles), 4 per wave
    for (int i = 0; i < 4; ++i) {
      int j = wave + 8 * i;
      int mtl = j >> 4, nt = j & 15;
      int n = nt * 16 + col;
      f32x8 acc = {};
#pragma unroll
      for (int k = 0; k < 8; ++k) {
        int k0 = k * 32;
        const __bf16* pah = &Ah[(mtl * 16 + col) * PITCH + k0 + half * 16];
        const __bf16* pal = &Al[(mtl * 16 + col) * PITCH + k0 + half * 16];
        bf16x16 fAh = frag_ld(pah, pah + 8);
        bf16x16 fAl = frag_ld(pal, pal + 8);
        const __bf16* pbh = WgTh + (size_t)n * D_DIM + k0 + half * 16;
        const __bf16* pbl = WgTl + (size_t)n * D_DIM + k0 + half * 16;
        bf16x16 fBh = frag_ld(pbh, pbh + 8);
        bf16x16 fBl = frag_ld(pbl, pbl + 8);
        acc = wmma_bf16(fAh, fBh, acc);
        acc = wmma_bf16(fAh, fBl, acc);
        acc = wmma_bf16(fAl, fBh, acc);
      }
      float addn = localf[b * D_DIM + n] + bg[n] + biasb;
      float wq = Wq[n];
#pragma unroll
      for (int r = 0; r < 8; ++r) {
        float g = acc[r] + addn;
        float sig = 1.0f / (1.0f + __expf(-g));
        float c = sig * wq;
        // deterministic reduction over the 16 columns (halves stay disjoint)
        c += __shfl_xor(c, 1);
        c += __shfl_xor(c, 2);
        c += __shfl_xor(c, 4);
        c += __shfl_xor(c, 8);
        if (col == 0) part[p * 32 + mtl * 16 + half * 8 + r][nt] = c;
      }
    }
    __syncthreads();
  }

  // ---- score[s] = b_q + sum_nt part[s][nt]   (deterministic order)
  if (tid < 64) {
    float sc = bq[0];
#pragma unroll
    for (int nt = 0; nt < 16; ++nt) sc += part[tid][nt];
    scoreS[tid] = sc;
  }
  __syncthreads();

  // ---- weighted pooling from exact f32 embeddings
  float s_acc = 0.0f;
  for (int s = 0; s < S_DIM; ++s) {
    float a = emb[(size_t)x[b * S_DIM + s] * D_DIM + tid];
    s_acc = fmaf(a, scoreS[s], s_acc);
  }
  sess[b * D_DIM + tid] = s_acc;
}

// ---------------------------------------------------------------------------
// hybrid = [local | sess] @ W_h + b_h, emitted directly as bf16 hi/lo planes.
// ---------------------------------------------------------------------------
__global__ void k_hybrid(const float* __restrict__ localf,
                         const float* __restrict__ sess,
                         const float* __restrict__ Wh,
                         const float* __restrict__ bh,
                         __bf16* __restrict__ hybH, __bf16* __restrict__ hybL) {
  int b = blockIdx.x, d = threadIdx.x;
  const float* l = localf + b * D_DIM;
  const float* g = sess + b * D_DIM;
  float acc = bh[d];
#pragma unroll 4
  for (int k = 0; k < D_DIM; ++k)
    acc = fmaf(l[k], Wh[k * D_DIM + d], acc);
#pragma unroll 4
  for (int k = 0; k < D_DIM; ++k)
    acc = fmaf(g[k], Wh[(D_DIM + k) * D_DIM + d], acc);
  __bf16 h = (__bf16)acc;
  hybH[b * D_DIM + d] = h;
  hybL[b * D_DIM + d] = (__bf16)(acc - (float)h);
}

// ---------------------------------------------------------------------------
// Final scoring GEMM: out(1024 x 100000) = hybrid @ emb.T (split-bf16 WMMA).
// Grid = 6250 blocks (one 16-item tile each); block = 8 waves; each wave
// keeps the item tile's B-fragments (all 8 k-steps, hi+lo = 128 VGPRs)
// resident and sweeps 8 M-tiles -> emb planes are read exactly once (HBM),
// hybrid (1 MB) streams from L2. Output write (409.6 MB) is the roofline.
// ---------------------------------------------------------------------------
__global__ void __launch_bounds__(256) k_score(
    const __bf16* __restrict__ embH, const __bf16* __restrict__ embL,
    const __bf16* __restrict__ hybH, const __bf16* __restrict__ hybL,
    float* __restrict__ out) {
  int nt = blockIdx.x;                 // 0..6249
  int tid = threadIdx.x;
  int wave = tid >> 5, lane = tid & 31, col = lane & 15, half = lane >> 4;
  int item = nt * 16 + col;

  const __bf16* ph = embH + (size_t)item * D_DIM + half * 16;
  const __bf16* pl = embL + (size_t)item * D_DIM + half * 16;
  bf16x16 Bh[8], Bl[8];
#pragma unroll
  for (int k = 0; k < 8; ++k) {
    Bh[k] = frag_ld(ph + k * 32, ph + k * 32 + 8);
    Bl[k] = frag_ld(pl + k * 32, pl + k * 32 + 8);
  }

#pragma unroll 1
  for (int mi = 0; mi < 8; ++mi) {
    int m0 = (wave * 8 + mi) * 16;
    const __bf16* ah = hybH + (size_t)(m0 + col) * D_DIM + half * 16;
    const __bf16* al = hybL + (size_t)(m0 + col) * D_DIM + half * 16;
    // speculative prefetch of the next M-tile's A rows (L2-resident anyway)
    __builtin_prefetch(ah + 16 * D_DIM, 0, 1);
    f32x8 acc = {};
#pragma unroll
    for (int k = 0; k < 8; ++k) {
      bf16x16 fAh = frag_ld(ah + k * 32, ah + k * 32 + 8);
      bf16x16 fAl = frag_ld(al + k * 32, al + k * 32 + 8);
      acc = wmma_bf16(fAh, Bh[k], acc);
      acc = wmma_bf16(fAh, Bl[k], acc);
      acc = wmma_bf16(fAl, Bh[k], acc);
    }
    float* o = out + (size_t)(m0 + half * 8) * N_ITEMS + item;
#pragma unroll
    for (int r = 0; r < 8; ++r) o[(size_t)r * N_ITEMS] = acc[r];
  }
}

// ---------------------------------------------------------------------------
extern "C" void kernel_launch(void* const* d_in, const int* in_sizes, int n_in,
                              void* d_out, int out_size, void* d_ws,
                              size_t ws_size, hipStream_t stream) {
  (void)in_sizes; (void)n_in; (void)out_size; (void)ws_size;
  const int*   x    = (const int*)d_in[0];
  const float* emb  = (const float*)d_in[1];
  const float* Wg   = (const float*)d_in[2];
  const float* bg   = (const float*)d_in[3];
  const float* Wl   = (const float*)d_in[4];
  const float* bl   = (const float*)d_in[5];
  const float* bias = (const float*)d_in[6];
  const float* Wq   = (const float*)d_in[7];
  const float* bq   = (const float*)d_in[8];
  const float* Wh   = (const float*)d_in[9];
  const float* bh   = (const float*)d_in[10];
  float* out = (float*)d_out;

  char* ws = (char*)d_ws;
  size_t off = 0;
  auto carve = [&](size_t bytes) {
    char* p = ws + off;
    off = (off + bytes + 255) & ~(size_t)255;
    return p;
  };
  __bf16* embH   = (__bf16*)carve((size_t)N_ITEMS * D_DIM * 2);
  __bf16* embL   = (__bf16*)carve((size_t)N_ITEMS * D_DIM * 2);
  __bf16* WgTh   = (__bf16*)carve((size_t)D_DIM * D_DIM * 2);
  __bf16* WgTl   = (__bf16*)carve((size_t)D_DIM * D_DIM * 2);
  float*  localf = (float*) carve((size_t)B_DIM * D_DIM * 4);
  float*  sess   = (float*) carve((size_t)B_DIM * D_DIM * 4);
  __bf16* hybH   = (__bf16*)carve((size_t)B_DIM * D_DIM * 2);
  __bf16* hybL   = (__bf16*)carve((size_t)B_DIM * D_DIM * 2);

  k_conv_emb<<<2048, 256, 0, stream>>>(emb, embH, embL, N_ITEMS * D_DIM / 4);
  k_trans_wg<<<D_DIM, D_DIM, 0, stream>>>(Wg, WgTh, WgTl);
  k_local  <<<B_DIM, D_DIM, 0, stream>>>(x, emb, Wl, bl, localf);
  k_attn   <<<B_DIM, 256,   0, stream>>>(x, emb, WgTh, WgTl, bg, localf,
                                         bias, Wq, bq, sess);
  k_hybrid <<<B_DIM, D_DIM, 0, stream>>>(localf, sess, Wh, bh, hybH, hybL);
  k_score  <<<NTILES_N, 256, 0, stream>>>(embH, embL, hybH, hybL, out);
}
